// EmergentResonator_77661598646296
// MI455X (gfx1250) — matrix operand order repaired and checked by reading, not compiled, not run on hardware
//
#include <hip/hip_runtime.h>

// ---------------------------------------------------------------------------
// EmergentResonator on MI455X (gfx1250, wave32, WMMA bf16 16x16x32)
//
// v3: same structure as v2 (32 batch rows/block, register-resident cur &
// potential, LDS bf16 spikes, shfl+LDS LayerNorm, zero-padded params,
// hoisted per-lane neuron constants) but sigmoids now use the hardware
// v_rcp_f32 (__builtin_amdgcn_rcpf) instead of the IEEE divide expansion,
// removing ~1000 static VALU ops from the 80x sequential elementwise stage.
// ---------------------------------------------------------------------------

#define NETN   360
#define NETP   384          // padded to 24 x 16
#define NTIL   24           // N tiles of 16
#define KTIL   12           // K tiles of 32
#define TSTEPS 64
#define INPF   28
#define BROWS  32           // batch rows per block (2 M-tiles of 16)
#define NCLS   10
#define LEAKF  0.95f
#define LNEPS  1e-5f

typedef __attribute__((ext_vector_type(16))) __bf16 v16bf;
typedef __attribute__((ext_vector_type(8)))  float  v8f;

union AF { v16bf v; unsigned int u[8]; };

// workspace layout (bytes)
static constexpr size_t WS_WREC  = 0;                  // 24*12*512*2 = 294912
static constexpr size_t WS_WIN   = 294912;             // 24*512*2    = 24576
static constexpr size_t WS_WGATE = 294912 + 24576;     // 24576
static constexpr size_t WS_PAR   = 294912 + 49152;     // 8*384*4     = 12288

__device__ __forceinline__ unsigned short f2bf(float f) {
    __bf16 b = (__bf16)f;
    unsigned short u;
    __builtin_memcpy(&u, &b, 2);
    return u;
}
__device__ __forceinline__ unsigned int pack2bf(float a, float b) {
    return (unsigned int)f2bf(a) | ((unsigned int)f2bf(b) << 16);
}
__device__ __forceinline__ float bf2f(unsigned short h) {
    unsigned int u = ((unsigned int)h) << 16;
    float f; __builtin_memcpy(&f, &u, 4);
    return f;
}
// fast sigmoid: v_exp_f32 + v_add_f32 + v_rcp_f32 (no IEEE div expansion)
__device__ __forceinline__ float sigm(float x) {
    return __builtin_amdgcn_rcpf(1.0f + __expf(-x));
}

// --- pack W_rec into B-fragment-major bf16: B[k][n] = W_rec[n][k] -----------
// frag element (lane, slot): n_in = lane&15, k_in = slot + 16*(lane>>4)
__global__ void pack_wrec(const float* __restrict__ W, unsigned short* __restrict__ o) {
    int idx = blockIdx.x * 256 + threadIdx.x;
    if (idx >= NTIL * KTIL * 512) return;
    int nt   = idx / (KTIL * 512);
    int rem  = idx % (KTIL * 512);
    int kt   = rem / 512;
    int e    = rem % 512;
    int lane = e / 16, slot = e % 16;
    int n = nt * 16 + (lane & 15);
    int k = kt * 32 + slot + 16 * (lane >> 4);
    float v = (n < NETN && k < NETN) ? W[n * NETN + k] : 0.0f;
    o[idx] = f2bf(v);
}

// --- pack W_in / W_gate (NET x 28, K padded to 32), single K tile -----------
__global__ void pack_wio(const float* __restrict__ W, unsigned short* __restrict__ o) {
    int idx = blockIdx.x * 256 + threadIdx.x;
    if (idx >= NTIL * 512) return;
    int nt   = idx / 512;
    int e    = idx % 512;
    int lane = e / 16, slot = e % 16;
    int n = nt * 16 + (lane & 15);
    int k = slot + 16 * (lane >> 4);
    float v = (n < NETN && k < INPF) ? W[n * INPF + k] : 0.0f;
    o[idx] = f2bf(v);
}

// --- zero-padded parameter vectors [8][NETP] --------------------------------
// 0:b_in 1:b_gate 2:gamma 3:beta 4:threshold 5:intrinsic 6:|steepness| 7:reset
__global__ void pack_params(const float* __restrict__ b_in,
                            const float* __restrict__ b_gate,
                            const float* __restrict__ gamma,
                            const float* __restrict__ beta,
                            const float* __restrict__ thr,
                            const float* __restrict__ intr,
                            const float* __restrict__ steep,
                            const float* __restrict__ rscale,
                            float* __restrict__ o) {
    int idx = blockIdx.x * 256 + threadIdx.x;
    if (idx >= 8 * NETP) return;
    int v = idx / NETP, n = idx % NETP;
    float val = 0.0f;
    if (n < NETN) {
        switch (v) {
            case 0: val = b_in[n];         break;
            case 1: val = b_gate[n];       break;
            case 2: val = gamma[n];        break;
            case 3: val = beta[n];         break;
            case 4: val = thr[n];          break;
            case 5: val = intr[n];         break;
            case 6: val = fabsf(steep[n]); break;
            case 7: val = rscale[n];       break;
        }
    }
    o[idx] = val;
}

// ---------------------------------------------------------------------------
__launch_bounds__(256)
__global__ void resonator_main(const float* __restrict__ x,
                               const int*   __restrict__ rsteps,
                               const float* __restrict__ parP,
                               const float* __restrict__ Wcls,
                               const float* __restrict__ bcls,
                               const unsigned short* __restrict__ wrecF,
                               const unsigned short* __restrict__ winF,
                               const unsigned short* __restrict__ wgateF,
                               float* __restrict__ out) {
    __shared__ unsigned short spk[BROWS][NETP];   // 24576 B
    __shared__ float          xt[BROWS * 32];     //  4096 B
    __shared__ float          red[8][BROWS][2];   //  2048 B
    __shared__ float          fin[BROWS][2];      //   256 B

    const int tid  = threadIdx.x;
    const int wave = tid >> 5;
    const int lane = tid & 31;
    const int half = lane >> 4;
    const int mrow = lane & 15;
    const int b0   = blockIdx.x * BROWS;

    // hoist per-lane neuron params (columns owned by this lane are fixed)
    float p_bi[3], p_bg[3], p_ga[3], p_be[3], p_th[3], p_in[3], p_st[3], p_rs[3];
    #pragma unroll
    for (int j = 0; j < 3; ++j) {
        int n = (wave + 8 * j) * 16 + mrow;
        p_bi[j] = parP[0 * NETP + n];
        p_bg[j] = parP[1 * NETP + n];
        p_ga[j] = parP[2 * NETP + n];
        p_be[j] = parP[3 * NETP + n];
        p_th[j] = parP[4 * NETP + n];
        p_in[j] = parP[5 * NETP + n];
        p_st[j] = parP[6 * NETP + n];
        p_rs[j] = parP[7 * NETP + n];
    }

    // init spikes
    for (int i = tid; i < BROWS * NETP; i += 256)
        (&spk[0][0])[i] = 0;

    v8f pot[2][3];
    {
        v8f z = {};
        #pragma unroll
        for (int mi = 0; mi < 2; ++mi)
            #pragma unroll
            for (int j = 0; j < 3; ++j)
                pot[mi][j] = z;
    }

    const int total = TSTEPS + rsteps[0];

    for (int t = 0; t < total; ++t) {
        __syncthreads();   // spikes from previous step are final

        v8f acc[2][3];

        if (t < TSTEPS) {
            // stage x[:, t, :] (32 x 28 -> padded 32) to LDS
            for (int i = tid; i < BROWS * 32; i += 256) {
                int r = i >> 5, c = i & 31;
                xt[i] = (c < INPF)
                          ? x[((size_t)(b0 + r) * TSTEPS + (size_t)t) * INPF + c]
                          : 0.0f;
            }
            __syncthreads();

            AF xf[2];
            #pragma unroll
            for (int mi = 0; mi < 2; ++mi)
                #pragma unroll
                for (int p = 0; p < 8; ++p) {
                    int kin = 2 * p + 16 * half;
                    int base = (mi * 16 + mrow) * 32 + kin;
                    xf[mi].u[p] = pack2bf(xt[base], xt[base + 1]);
                }

            v8f zero = {};
            #pragma unroll
            for (int j = 0; j < 3; ++j) {
                int nt = wave + 8 * j;
                v16bf bw = *(const v16bf*)(winF   + (size_t)nt * 512 + (size_t)lane * 16);
                v16bf bg = *(const v16bf*)(wgateF + (size_t)nt * 512 + (size_t)lane * 16);
                #pragma unroll
                for (int mi = 0; mi < 2; ++mi) {
                    v8f ia = __builtin_amdgcn_wmma_f32_16x16x32_bf16(
                                 false, xf[mi].v, false, bw, (short)0, zero, false, false);
                    v8f ig = __builtin_amdgcn_wmma_f32_16x16x32_bf16(
                                 false, xf[mi].v, false, bg, (short)0, zero, false, false);
                    #pragma unroll
                    for (int i2 = 0; i2 < 8; ++i2)
                        acc[mi][j][i2] = (ia[i2] + p_bi[j]) * sigm(ig[i2] + p_bg[j]);
                }
            }
        } else {
            v8f zero = {};
            #pragma unroll
            for (int mi = 0; mi < 2; ++mi)
                #pragma unroll
                for (int j = 0; j < 3; ++j)
                    acc[mi][j] = zero;
        }

        // recurrent GEMM: 12 K-chunks x 3 N-tiles x 2 M-tiles = 72 WMMAs
        #pragma unroll
        for (int kt = 0; kt < KTIL; ++kt) {
            AF af[2];
            const int kb = kt * 32;
            #pragma unroll
            for (int mi = 0; mi < 2; ++mi)
                #pragma unroll
                for (int p = 0; p < 8; ++p) {
                    int kin = 2 * p + 16 * half;
                    af[mi].u[p] = *(const unsigned int*)&spk[mi * 16 + mrow][kb + kin];
                }
            #pragma unroll
            for (int j = 0; j < 3; ++j) {
                int nt = wave + 8 * j;
                v16bf bw = *(const v16bf*)(wrecF +
                               ((size_t)(nt * KTIL + kt) * 32 + (size_t)lane) * 16);
                acc[0][j] = __builtin_amdgcn_wmma_f32_16x16x32_bf16(
                                false, af[0].v, false, bw, (short)0, acc[0][j], false, false);
                acc[1][j] = __builtin_amdgcn_wmma_f32_16x16x32_bf16(
                                false, af[1].v, false, bw, (short)0, acc[1][j], false, false);
            }
        }

        // LayerNorm stage 1: per-wave row partial sums (shfl within 16-lane group)
        #pragma unroll
        for (int mi = 0; mi < 2; ++mi)
            #pragma unroll
            for (int i2 = 0; i2 < 8; ++i2) {
                float a0 = acc[mi][0][i2], a1 = acc[mi][1][i2], a2 = acc[mi][2][i2];
                float s1 = a0 + a1 + a2;
                float s2 = a0 * a0 + a1 * a1 + a2 * a2;
                #pragma unroll
                for (int off = 1; off < 16; off <<= 1) {
                    s1 += __shfl_xor(s1, off, 32);
                    s2 += __shfl_xor(s2, off, 32);
                }
                if (mrow == 0) {
                    int m = i2 + 8 * half + 16 * mi;
                    red[wave][m][0] = s1;
                    red[wave][m][1] = s2;
                }
            }
        __syncthreads();

        // LayerNorm stage 2: reduce 8 wave partials, broadcast mu / rstd
        if (tid < BROWS) {
            float s1 = 0.0f, s2 = 0.0f;
            #pragma unroll
            for (int w = 0; w < 8; ++w) {
                s1 += red[w][tid][0];
                s2 += red[w][tid][1];
            }
            float mu  = s1 * (1.0f / NETN);
            float var = s2 * (1.0f / NETN) - mu * mu;   // pad cols are exact 0
            fin[tid][0] = mu;
            fin[tid][1] = rsqrtf(var + LNEPS);
        }
        __syncthreads();

        // PulseTGAU neuron step on register-resident cur/potential
        #pragma unroll
        for (int mi = 0; mi < 2; ++mi)
            #pragma unroll
            for (int j = 0; j < 3; ++j) {
                int n = (wave + 8 * j) * 16 + mrow;
                #pragma unroll
                for (int i2 = 0; i2 < 8; ++i2) {
                    int m      = i2 + 8 * half + 16 * mi;
                    float mu   = fin[m][0];
                    float rstd = fin[m][1];
                    float c    = acc[mi][j][i2];
                    float h    = (c - mu) * rstd * p_ga[j] + p_be[j];
                    float p    = pot[mi][j][i2] * LEAKF + h + p_in[j];
                    float gp   = p - p_th[j];
                    float sg   = sigm(gp * p_st[j]);
                    float o    = gp * sigm(gp) * sg;        // silu(gp) * sg
                    pot[mi][j][i2] = p - p_rs[j] * (gp * sg);
                    spk[m][n]  = f2bf(o);
                }
            }
    }

    __syncthreads();

    // classifier: out = spikes @ W_cls^T + b_cls   (32 rows x 10 classes)
    for (int idx = tid; idx < BROWS * NCLS; idx += 256) {
        int r = idx / NCLS, c = idx % NCLS;
        float s = bcls[c];
        for (int n = 0; n < NETN; ++n)
            s += bf2f(spk[r][n]) * Wcls[c * NETN + n];
        out[(size_t)(b0 + r) * NCLS + c] = s;
    }
}

// ---------------------------------------------------------------------------
extern "C" void kernel_launch(void* const* d_in, const int* in_sizes, int n_in,
                              void* d_out, int out_size, void* d_ws, size_t ws_size,
                              hipStream_t stream) {
    const float* x      = (const float*)d_in[0];
    const int*   rsteps = (const int*)  d_in[1];
    const float* W_in   = (const float*)d_in[2];
    const float* b_in   = (const float*)d_in[3];
    const float* W_gate = (const float*)d_in[4];
    const float* b_gate = (const float*)d_in[5];
    const float* W_rec  = (const float*)d_in[6];
    const float* gamma  = (const float*)d_in[7];
    const float* beta   = (const float*)d_in[8];
    const float* thr    = (const float*)d_in[9];
    const float* intr   = (const float*)d_in[10];
    const float* steep  = (const float*)d_in[11];
    const float* rscale = (const float*)d_in[12];
    const float* Wcls   = (const float*)d_in[13];
    const float* bcls   = (const float*)d_in[14];
    float* out = (float*)d_out;

    unsigned short* wrecF  = (unsigned short*)((char*)d_ws + WS_WREC);
    unsigned short* winF   = (unsigned short*)((char*)d_ws + WS_WIN);
    unsigned short* wgateF = (unsigned short*)((char*)d_ws + WS_WGATE);
    float*          parP   = (float*)        ((char*)d_ws + WS_PAR);

    const int B = in_sizes[0] / (TSTEPS * INPF);   // 4096

    pack_wrec  <<<(NTIL * KTIL * 512 + 255) / 256, 256, 0, stream>>>(W_rec, wrecF);
    pack_wio   <<<(NTIL * 512 + 255) / 256,        256, 0, stream>>>(W_in,   winF);
    pack_wio   <<<(NTIL * 512 + 255) / 256,        256, 0, stream>>>(W_gate, wgateF);
    pack_params<<<(8 * NETP + 255) / 256,          256, 0, stream>>>(
        b_in, b_gate, gamma, beta, thr, intr, steep, rscale, parP);

    resonator_main<<<B / BROWS, 256, 0, stream>>>(
        x, rsteps, parP, Wcls, bcls, wrecF, winF, wgateF, out);
}
